// GCN_30743375904983
// MI455X (gfx1250) — compile-verified, matrix operand order checked by reading, not run on hardware
//
#include <hip/hip_runtime.h>
#include <hip/hip_bf16.h>
#include <cstdint>
#include <cstddef>

// ---------------------------------------------------------------------------
// GCN 2-layer forward for MI455X (gfx1250, wave32, WMMA).
//   h  = relu( (X @ W1) scattered over edges + b1 )
//   o  = log_softmax( (h @ W2) scattered over edges + b2 )
// GEMMs: v_wmma_f32_16x16x32_f16, B pre-packed in fragment-native layout so
// each lane loads its 16 halves as one contiguous 32B vector (2x b128).
// ---------------------------------------------------------------------------

typedef __attribute__((ext_vector_type(16))) _Float16 v16h;
typedef __attribute__((ext_vector_type(8)))  float    v8f;

// A-fragment: 16x32 f16 from row-major f32, vectorized float4 loads + cvt.
// ISA layout: lanes 0-15 hold row M=lane, K = {0..7} in v0-3, {16..23} in v4-7;
// lanes 16-31 hold K = {8..15}, {24..31}.
__device__ __forceinline__ v16h load_a_frag(const float* __restrict__ A, int lda,
                                            int row0, int k0, int lane, int do_relu) {
  const int hi = lane >> 4;
  const int m  = lane & 15;
  const float* p = A + (size_t)(row0 + m) * lda + k0 + hi * 8;
  float4 u0 = *reinterpret_cast<const float4*>(p);
  float4 u1 = *reinterpret_cast<const float4*>(p + 4);
  float4 u2 = *reinterpret_cast<const float4*>(p + 16);
  float4 u3 = *reinterpret_cast<const float4*>(p + 20);
  if (do_relu) {
    u0.x = fmaxf(u0.x, 0.f); u0.y = fmaxf(u0.y, 0.f); u0.z = fmaxf(u0.z, 0.f); u0.w = fmaxf(u0.w, 0.f);
    u1.x = fmaxf(u1.x, 0.f); u1.y = fmaxf(u1.y, 0.f); u1.z = fmaxf(u1.z, 0.f); u1.w = fmaxf(u1.w, 0.f);
    u2.x = fmaxf(u2.x, 0.f); u2.y = fmaxf(u2.y, 0.f); u2.z = fmaxf(u2.z, 0.f); u2.w = fmaxf(u2.w, 0.f);
    u3.x = fmaxf(u3.x, 0.f); u3.y = fmaxf(u3.y, 0.f); u3.z = fmaxf(u3.z, 0.f); u3.w = fmaxf(u3.w, 0.f);
  }
  v16h a;
  a[0]  = (_Float16)u0.x; a[1]  = (_Float16)u0.y; a[2]  = (_Float16)u0.z; a[3]  = (_Float16)u0.w;
  a[4]  = (_Float16)u1.x; a[5]  = (_Float16)u1.y; a[6]  = (_Float16)u1.z; a[7]  = (_Float16)u1.w;
  a[8]  = (_Float16)u2.x; a[9]  = (_Float16)u2.y; a[10] = (_Float16)u2.z; a[11] = (_Float16)u2.w;
  a[12] = (_Float16)u3.x; a[13] = (_Float16)u3.y; a[14] = (_Float16)u3.z; a[15] = (_Float16)u3.w;
  return a;
}

// One wave computes one 16-row tile across NT column tiles (A reused NT times).
// Bp is fragment-packed: fragment (kt, t) occupies 32 lanes x 16 halves,
// lane-contiguous:  Bp[ ((kt*NT + t)*32 + lane)*16 + j ].
template <int NT, int RELU>
__global__ void gemm_wmma_kernel(const float* __restrict__ A,
                                 const _Float16* __restrict__ Bp,
                                 float* __restrict__ C,
                                 int ntileM, int K, int lda, int ldc, int ncols) {
  const int wid  = (int)((blockIdx.x * blockDim.x + threadIdx.x) >> 5);
  const int lane = (int)(threadIdx.x & 31);
  if (wid >= ntileM) return;
  const int row0 = wid * 16;
  const v16h* __restrict__ Bfrag = reinterpret_cast<const v16h*>(Bp);

  v8f acc[NT];
#pragma unroll
  for (int t = 0; t < NT; ++t) acc[t] = (v8f){0.f,0.f,0.f,0.f,0.f,0.f,0.f,0.f};

  for (int k0 = 0; k0 < K; k0 += 32) {
    const v16h a = load_a_frag(A, lda, row0, k0, lane, RELU);
    const size_t fbase = ((size_t)(k0 >> 5) * NT) * 32 + lane;
#pragma unroll
    for (int t = 0; t < NT; ++t) {
      const v16h b = Bfrag[fbase + (size_t)t * 32];  // one 32B contiguous load
      acc[t] = __builtin_amdgcn_wmma_f32_16x16x32_f16(
          /*neg_a=*/false, a, /*neg_b=*/false, b,
          /*c_mod=*/(short)0, acc[t], /*reuse_a=*/false, /*reuse_b=*/false);
    }
  }

  // C/D layout: lanes 0-15: N=lane, M=vgpr r; lanes 16-31: N=lane-16, M=8+r.
  const int hi = lane >> 4;
  const int n  = lane & 15;
#pragma unroll
  for (int t = 0; t < NT; ++t) {
    const int col = t * 16 + n;
    if (col < ncols) {
#pragma unroll
      for (int r = 0; r < 8; ++r)
        C[(size_t)(row0 + hi * 8 + r) * ldc + col] = acc[t][r];
    }
  }
}

// Pack f32 weights W[K x ncol] into WMMA-B fragment-native f16 layout.
// Fragment (kt, t): lane hi*16+n, element j -> W[kt*32 + hi*8 + (j&7) + (j>=8?16:0)][t*16+n]
__global__ void pack_b_kernel(const float* __restrict__ W, _Float16* __restrict__ Bp,
                              int K, int ncol, int ntn) {
  const int i = blockIdx.x * blockDim.x + threadIdx.x;
  const int total = (K >> 5) * ntn * 512;   // 32 lanes * 16 halves per fragment
  if (i >= total) return;
  const int j    = i & 15;
  const int lane = (i >> 4) & 31;
  const int frag = i >> 9;
  const int t  = frag % ntn;
  const int kt = frag / ntn;
  const int hi = lane >> 4;
  const int n  = lane & 15;
  const int k   = kt * 32 + hi * 8 + (j & 7) + ((j & 8) << 1);
  const int col = t * 16 + n;
  _Float16 v = (_Float16)0.0f;
  if (col < ncol) v = (_Float16)W[(size_t)k * ncol + col];
  Bp[i] = v;
}

// dst[i] = b[i % nfeat]  (broadcast bias as the scatter-add init value)
__global__ void bias_init_kernel(float* __restrict__ dst, const float* __restrict__ b,
                                 int nfeat, size_t total) {
  size_t i = (size_t)blockIdx.x * blockDim.x + threadIdx.x;
  const size_t stride = (size_t)gridDim.x * blockDim.x;
  for (; i < total; i += stride) dst[i] = b[(int)(i % (size_t)nfeat)];
}

// Thread per (edge, 4-feature chunk): agg[dst] += w_e * sup[src].
template <int NF4>
__global__ void scatter_kernel(const float* __restrict__ sup,
                               const long long* __restrict__ src,
                               const long long* __restrict__ dst,
                               const float* __restrict__ ew,
                               float* __restrict__ agg,
                               long long E) {
  const long long tid = (long long)blockIdx.x * blockDim.x + threadIdx.x;
  if (tid >= E * NF4) return;
  const long long e  = tid / NF4;          // div by constant -> mul/shift
  const int       f  = (int)(tid - e * NF4) * 4;
  const long long s = src[e];
  const long long d = dst[e];
  const float w = ew[e];
  const float4 v = *reinterpret_cast<const float4*>(sup + (size_t)s * (NF4 * 4) + f);
  float* ap = agg + (size_t)d * (NF4 * 4) + f;
  atomicAdd(ap + 0, w * v.x);
  atomicAdd(ap + 1, w * v.y);
  atomicAdd(ap + 2, w * v.z);
  atomicAdd(ap + 3, w * v.w);
}

// In-place log_softmax over rows of nfeat (<=64) columns, one wave per row.
__global__ void log_softmax_kernel(float* __restrict__ out, int nrows, int nfeat) {
  const int row  = (int)((blockIdx.x * blockDim.x + threadIdx.x) >> 5);
  const int lane = (int)(threadIdx.x & 31);
  if (row >= nrows) return;
  float* r = out + (size_t)row * nfeat;
  const float NEG = -3.0e38f;
  float v0 = (lane < nfeat) ? r[lane] : NEG;
  float v1 = (lane + 32 < nfeat) ? r[lane + 32] : NEG;
  float m = fmaxf(v0, v1);
#pragma unroll
  for (int o = 16; o > 0; o >>= 1) m = fmaxf(m, __shfl_xor(m, o, 32));
  float s = 0.0f;
  if (lane < nfeat)      s += __expf(v0 - m);
  if (lane + 32 < nfeat) s += __expf(v1 - m);
#pragma unroll
  for (int o = 16; o > 0; o >>= 1) s += __shfl_xor(s, o, 32);
  const float lse = m + __logf(s);
  if (lane < nfeat)      r[lane]      = v0 - lse;
  if (lane + 32 < nfeat) r[lane + 32] = v1 - lse;
}

extern "C" void kernel_launch(void* const* d_in, const int* in_sizes, int n_in,
                              void* d_out, int out_size, void* d_ws, size_t ws_size,
                              hipStream_t stream) {
  const float*     x  = (const float*)d_in[0];
  const long long* ei = (const long long*)d_in[1];   // int64 [2, E]
  const float*     ew = (const float*)d_in[2];
  const float*     W1 = (const float*)d_in[3];
  const float*     b1 = (const float*)d_in[4];
  const float*     W2 = (const float*)d_in[5];
  const float*     b2 = (const float*)d_in[6];
  float* out = (float*)d_out;

  const int F0 = 512, F1 = 128, F2 = 40;
  const int NT1 = F1 / 16;        // 8 col tiles, layer 1
  const int NT2 = 3;              // 3 col tiles (48 padded), layer 2
  const int N = in_sizes[0] / F0;              // 100000
  const long long E = (long long)in_sizes[2];  // 3200000
  const long long* src = ei;
  const long long* dst = ei + E;

  // Workspace layout
  char* ws = (char*)d_ws;
  _Float16* B1p = (_Float16*)ws;  ws += (size_t)(F0 / 32) * NT1 * 512 * sizeof(_Float16); // 128 KB
  _Float16* B2p = (_Float16*)ws;  ws += (size_t)(F1 / 32) * NT2 * 512 * sizeof(_Float16); //  12 KB
  float* sup1 = (float*)ws;       ws += (size_t)N * F1 * sizeof(float);                   // 51.2 MB
  float* h1   = (float*)ws;       ws += (size_t)N * F1 * sizeof(float);                   // 51.2 MB
  float* sup2 = (float*)ws;       ws += (size_t)N * F2 * sizeof(float);                   // 16 MB

  // 1) Weights -> fragment-packed f16
  {
    int t1 = (F0 / 32) * NT1 * 512;
    pack_b_kernel<<<(t1 + 255) / 256, 256, 0, stream>>>(W1, B1p, F0, F1, NT1);
    int t2 = (F1 / 32) * NT2 * 512;
    pack_b_kernel<<<(t2 + 255) / 256, 256, 0, stream>>>(W2, B2p, F1, F2, NT2);
  }

  const int ntileM = N / 16;  // 6250

  // 2) support1 = X @ W1
  gemm_wmma_kernel<8, 0><<<(ntileM + 3) / 4, 128, 0, stream>>>(
      x, B1p, sup1, ntileM, F0, F0, F1, F1);

  // 3) h1 = b1; h1[dst] += w_e * support1[src]
  {
    size_t tot = (size_t)N * F1;
    bias_init_kernel<<<(int)((tot + 1023) / 1024), 256, 0, stream>>>(h1, b1, F1, tot);
    long long nt = E * 32;
    scatter_kernel<32><<<(int)((nt + 255) / 256), 256, 0, stream>>>(sup1, src, dst, ew, h1, E);
  }

  // 4) support2 = relu(h1) @ W2  (fused ReLU on A, guarded store to 40 cols)
  gemm_wmma_kernel<3, 1><<<(ntileM + 3) / 4, 128, 0, stream>>>(
      h1, B2p, sup2, ntileM, F1, F1, F2, F2);

  // 5) out = b2; out[dst] += w_e * support2[src]
  {
    size_t tot = (size_t)N * F2;
    bias_init_kernel<<<(int)((tot + 1023) / 1024), 256, 0, stream>>>(out, b2, F2, tot);
    long long nt = E * 10;
    scatter_kernel<10><<<(int)((nt + 255) / 256), 256, 0, stream>>>(sup2, src, dst, ew, out, E);
  }

  // 6) in-place log_softmax over 40 columns, one wave per row
  log_softmax_kernel<<<(N + 7) / 8, 256, 0, stream>>>(out, N, F2);
}
